// PointGroup_74406013436004
// MI455X (gfx1250) — compile-verified, matrix operand order checked by reading, not run on hardware
//
#include <hip/hip_runtime.h>
#include <hip/hip_bf16.h>

#define N_PTS   200000
#define TILES   12500              // N_PTS / 16
#define P_INST  64
#define NPAR    337
#define INV_N   (1.0f / 200000.0f)
#define BN_EPS  1e-5f

typedef __attribute__((ext_vector_type(16))) _Float16 v16h;
typedef __attribute__((ext_vector_type(8)))  float    v8f;

// ---------------- workspace layout (bytes) ----------------
#define WS_STATS 0                       // 3 layers * 64 f32 (sum[32], sumsq[32])
#define WS_W1A   1024                    // 64*32*16 f16 = 65536  (per-p A tiles)
#define WS_MASKA 66560                   // 3*2*32*16 f16 = 6144  (mask layer A tiles)
#define WS_WOUTA 72704                   // 32*16 f16 = 1024      (w_out A tile)
#define WS_B1C   73728                   // 64*16 f32 = 4096      (center-folded b1')
#define WS_W2PK  77824                   // 64*16 f32 = 4096
#define WS_B2PK  81920                   // 64 f32
#define WS_YA    98304                   // N*32 f16 = 12,800,000 (packed B-layout)
#define WS_YB    (98304 + 12800000)

// element e of a 16-bit WMMA A/B operand holds K = kmap(lane, e)
__device__ __forceinline__ int kmap(int lane, int e) {
  return e + ((e >= 8) ? 8 : 0) + ((lane >= 16) ? 8 : 0);
}

// recompute params[p][j] from LDS-resident e (controller output)
__device__ __forceinline__ float param_at(const float* sE, const float* ctrl_w,
                                          const float* ctrl_b, int p, int j) {
  float acc = ctrl_b[j];
#pragma unroll
  for (int d = 0; d < 16; ++d) acc += sE[p * 16 + d] * ctrl_w[d * NPAR + j];
  return acc;
}

// ================= setup: controller path + operand packing =================
__global__ void setup_kernel(const float* inst_feats, const float* centers,
                             const float* mask_w1, const float* mask_w2,
                             const float* mask_w3, const float* mask_w_out,
                             const float* emb_w, const float* emb_g, const float* emb_b,
                             const float* ctrl_w, const float* ctrl_b,
                             _Float16* w1A, _Float16* maskA, _Float16* woutA,
                             float* b1c, float* w2pk, float* b2pk, float* stats) {
  __shared__ float sY[P_INST * 16];
  __shared__ float sE[P_INST * 16];
  __shared__ float sA[16], sB[16];
  const int t = threadIdx.x;  // 256 threads

  // y = inst_feats @ emb_w
  for (int idx = t; idx < P_INST * 16; idx += 256) {
    int p = idx >> 4, d = idx & 15;
    float acc = 0.f;
    for (int k = 0; k < 32; ++k) acc += inst_feats[p * 32 + k] * emb_w[k * 16 + d];
    sY[idx] = acc;
  }
  __syncthreads();
  if (t < 16) {
    float s = 0.f, q = 0.f;
    for (int p = 0; p < P_INST; ++p) { float v = sY[p * 16 + t]; s += v; q += v * v; }
    float mu = s * (1.0f / 64.0f);
    float var = q * (1.0f / 64.0f) - mu * mu;
    float a = emb_g[t] * rsqrtf(var + BN_EPS);
    sA[t] = a; sB[t] = emb_b[t] - mu * a;
  }
  __syncthreads();
  for (int idx = t; idx < P_INST * 16; idx += 256) {
    int d = idx & 15;
    sE[idx] = fmaxf(sA[d] * sY[idx] + sB[d], 0.f);
  }
  __syncthreads();

  // per-instance A tiles: A[d][K] = w1[p][d][K] (K<19), WMMA 16-bit A layout
  for (int idx = t; idx < P_INST * 32 * 16; idx += 256) {
    int p = idx >> 9, lane = (idx >> 4) & 31, e = idx & 15;
    int Mr = lane & 15, K = kmap(lane, e);
    float v = 0.f;
    if (K < 19) v = param_at(sE, ctrl_w, ctrl_b, p, Mr * 19 + K);
    w1A[idx] = (_Float16)v;
  }
  // b1' (center-folded), w2, b2
  for (int idx = t; idx < P_INST * 16; idx += 256) {
    int p = idx >> 4, f = idx & 15;
    float b1 = param_at(sE, ctrl_w, ctrl_b, p, 320 + f);
    for (int c = 0; c < 3; ++c)
      b1 -= param_at(sE, ctrl_w, ctrl_b, p, f * 19 + c) * centers[p * 3 + c];
    b1c[idx]  = b1;
    w2pk[idx] = param_at(sE, ctrl_w, ctrl_b, p, 304 + f);
  }
  for (int p = t; p < P_INST; p += 256) b2pk[p] = param_at(sE, ctrl_w, ctrl_b, p, 336);

  // mask-MLP A tiles: A[Mg][K] = W[K][Mg], two 16-row tiles per 32x32 layer
  for (int idx = t; idx < 3 * 2 * 32 * 16; idx += 256) {
    int l = idx / 1024, r2 = idx & 1023;
    int tt = r2 >> 9, lane = (r2 >> 4) & 31, e = r2 & 15;
    int Mg = tt * 16 + (lane & 15), K = kmap(lane, e);
    const float* W = (l == 0) ? mask_w1 : ((l == 1) ? mask_w2 : mask_w3);
    maskA[idx] = (_Float16)W[K * 32 + Mg];
  }
  // w_out A tile: A[M][K] = w_out[K][M]  (w_out is 32x16)
  for (int idx = t; idx < 32 * 16; idx += 256) {
    int lane = (idx >> 4) & 31, e = idx & 15;
    int Mr = lane & 15, K = kmap(lane, e);
    woutA[idx] = (_Float16)mask_w_out[K * 16 + Mr];
  }
  for (int i = t; i < 192; i += 256) stats[i] = 0.f;
}

// ================= mask MLP layer: y = B @ W, stats for next BN =================
// mode 0: B from f32 feats (row-major N x 32);  mode 1: B from packed f16 + bn + relu
__global__ void mlp_layer_kernel(const float* feats, const _Float16* yIn,
                                 const _Float16* Apack, const float* statsPrev,
                                 const float* gPrev, const float* bPrev,
                                 _Float16* yOut, float* statsOut, int mode) {
  __shared__ float sStat[64];
  const int tid = threadIdx.x, lane = tid & 31, wib = tid >> 5;
  if (tid < 64) sStat[tid] = 0.f;
  __syncthreads();
  const int gw = blockIdx.x * (blockDim.x >> 5) + wib;
  const int nw = gridDim.x * (blockDim.x >> 5);
  const int h = lane >> 4;

  v16h a0 = *(const v16h*)(Apack + lane * 16);
  v16h a1 = *(const v16h*)(Apack + (32 + lane) * 16);

  float an[16], bb[16];
  if (mode) {
#pragma unroll
    for (int e = 0; e < 16; ++e) {
      int f = kmap(lane, e);
      float mu  = statsPrev[f] * INV_N;
      float var = statsPrev[32 + f] * INV_N - mu * mu;
      float a = gPrev[f] * rsqrtf(var + BN_EPS);
      an[e] = a; bb[e] = bPrev[f] - mu * a;
    }
  }

  float accS[16], accQ[16];
#pragma unroll
  for (int e = 0; e < 16; ++e) { accS[e] = 0.f; accQ[e] = 0.f; }

  for (int tIdx = gw; tIdx < TILES; tIdx += nw) {
    v16h b;
    if (mode == 0) {
      int n = tIdx * 16 + (lane & 15);
      const float4* rf4 = (const float4*)(feats + n * 32);
      float4 q0 = rf4[2 * h], q1 = rf4[2 * h + 1];
      float4 q2 = rf4[2 * h + 4], q3 = rf4[2 * h + 5];
      b[0]=(_Float16)q0.x;  b[1]=(_Float16)q0.y;  b[2]=(_Float16)q0.z;  b[3]=(_Float16)q0.w;
      b[4]=(_Float16)q1.x;  b[5]=(_Float16)q1.y;  b[6]=(_Float16)q1.z;  b[7]=(_Float16)q1.w;
      b[8]=(_Float16)q2.x;  b[9]=(_Float16)q2.y;  b[10]=(_Float16)q2.z; b[11]=(_Float16)q2.w;
      b[12]=(_Float16)q3.x; b[13]=(_Float16)q3.y; b[14]=(_Float16)q3.z; b[15]=(_Float16)q3.w;
    } else {
      v16h raw = *(const v16h*)(yIn + (tIdx * 32 + lane) * 16);
#pragma unroll
      for (int e = 0; e < 16; ++e)
        b[e] = (_Float16)fmaxf(an[e] * (float)raw[e] + bb[e], 0.f);
    }
    v8f c0 = {};
    v8f d0 = __builtin_amdgcn_wmma_f32_16x16x32_f16(false, a0, false, b, (short)0, c0, false, false);
    v8f d1 = __builtin_amdgcn_wmma_f32_16x16x32_f16(false, a1, false, b, (short)0, c0, false, false);

    v16h po;
#pragma unroll
    for (int e = 0; e < 8; ++e) {
      float v = d0[e];
      po[e] = (_Float16)v; accS[e] += v; accQ[e] += v * v;
    }
#pragma unroll
    for (int e = 8; e < 16; ++e) {
      float v = d1[e - 8];
      po[e] = (_Float16)v; accS[e] += v; accQ[e] += v * v;
    }
    *(v16h*)(yOut + (tIdx * 32 + lane) * 16) = po;
  }

#pragma unroll
  for (int e = 0; e < 16; ++e) {
    float s = accS[e], q = accQ[e];
#pragma unroll
    for (int off = 1; off < 16; off <<= 1) {
      s += __shfl_xor(s, off, 32);
      q += __shfl_xor(q, off, 32);
    }
    if ((lane & 15) == 0) {
      int f = kmap(lane, e);
      atomicAdd(&sStat[f], s);
      atomicAdd(&sStat[32 + f], q);
    }
  }
  __syncthreads();
  if (tid < 64) atomicAdd(&statsOut[tid], sStat[tid]);
}

// ======= aug: mask_feats = bn(y3)@w_out + b_out, fused with coords into B layout =======
__global__ void aug_kernel(const _Float16* yIn, const _Float16* woutA,
                           const float* statsPrev, const float* gPrev, const float* bPrev,
                           const float* bout, const float* coords, _Float16* augOut) {
  const int tid = threadIdx.x, lane = tid & 31, wib = tid >> 5;
  const int gw = blockIdx.x * (blockDim.x >> 5) + wib;
  const int nw = gridDim.x * (blockDim.x >> 5);
  const int h = lane >> 4;

  v16h a0 = *(const v16h*)(woutA + lane * 16);

  float an[16], bb[16];
#pragma unroll
  for (int e = 0; e < 16; ++e) {
    int f = kmap(lane, e);
    float mu  = statsPrev[f] * INV_N;
    float var = statsPrev[32 + f] * INV_N - mu * mu;
    float a = gPrev[f] * rsqrtf(var + BN_EPS);
    an[e] = a; bb[e] = bPrev[f] - mu * a;
  }
  float bo[8];
#pragma unroll
  for (int r = 0; r < 8; ++r) bo[r] = bout[r + 8 * h];

  for (int tIdx = gw; tIdx < TILES; tIdx += nw) {
    v16h raw = *(const v16h*)(yIn + (tIdx * 32 + lane) * 16);
    v16h b;
#pragma unroll
    for (int e = 0; e < 16; ++e)
      b[e] = (_Float16)fmaxf(an[e] * (float)raw[e] + bb[e], 0.f);

    v8f c0 = {};
    v8f d = __builtin_amdgcn_wmma_f32_16x16x32_f16(false, a0, false, b, (short)0, c0, false, false);

    float m[8], pm[8];
#pragma unroll
    for (int r = 0; r < 8; ++r) {
      m[r] = d[r] + bo[r];
      pm[r] = __shfl_xor(m[r], 16, 32);  // partner half holds the other 8 features
    }
    int n = tIdx * 16 + (lane & 15);
    float c3x = coords[n * 3 + 0], c3y = coords[n * 3 + 1], c3z = coords[n * 3 + 2];

    v16h po;
#pragma unroll
    for (int e = 0; e < 16; ++e) {
      int K = kmap(lane, e);
      float v;
      if (K == 0)      v = c3x;
      else if (K == 1) v = c3y;
      else if (K == 2) v = c3z;
      else if (K < 19) {
        int j = K - 3;
        v = ((j >> 3) == h) ? m[j & 7] : pm[j & 7];
      } else v = 0.f;
      po[e] = (_Float16)v;
    }
    *(v16h*)(augOut + (tIdx * 32 + lane) * 16) = po;
  }
}

// ================= heavy: logits[p][n] via batched WMMA =================
// Per-block: stage all 64 KB of per-instance A tiles into LDS with CDNA5
// async global->LDS copies (ASYNCcnt), then 8 waves x 64 instances of WMMA.
__global__ void heavy_kernel(const _Float16* augPk, const _Float16* w1A,
                             const float* b1c, const float* w2pk, const float* b2pk,
                             float* out) {
  __shared__ _Float16 sW1A[P_INST * 32 * 16];  // 64 KB
  const int tid = threadIdx.x, lane = tid & 31, wib = tid >> 5;

  // ---- async global->LDS staging: 256 threads x 16 x b128 = 64 KB ----
  {
    unsigned ldsBase = (unsigned)(uintptr_t)&sW1A[0];  // low 32 bits = LDS offset
    const char* gsrc = (const char*)w1A;
#pragma unroll
    for (int it = 0; it < 16; ++it) {
      unsigned off = (unsigned)((it * 256 + tid) * 16);
      unsigned ldsAddr = ldsBase + off;
      const char* ga = gsrc + off;
      asm volatile("global_load_async_to_lds_b128 %0, %1, off"
                   :: "v"(ldsAddr), "v"(ga) : "memory");
    }
  }
  asm volatile("s_wait_asynccnt 0" ::: "memory");
  __syncthreads();  // all threads staged before any wave reads LDS

  const int gw = blockIdx.x * (blockDim.x >> 5) + wib;
  if (gw < TILES) {  // wave-uniform guard; no barriers below
    const int h = lane >> 4;
    v16h b = *(const v16h*)(augPk + (gw * 32 + lane) * 16);
    const int n = gw * 16 + (lane & 15);

    for (int p = 0; p < P_INST; ++p) {
      v16h a = *(const v16h*)(sW1A + (p * 32 + lane) * 16);  // ds_load_b128 x2
      const float4* bc = (const float4*)(b1c + p * 16 + h * 8);
      float4 c01 = bc[0], c23 = bc[1];
      v8f c = {c01.x, c01.y, c01.z, c01.w, c23.x, c23.y, c23.z, c23.w};  // bias in C

      v8f d = __builtin_amdgcn_wmma_f32_16x16x32_f16(false, a, false, b, (short)0, c, false, false);

      const float4* wv = (const float4*)(w2pk + p * 16 + h * 8);
      float4 w01 = wv[0], w23 = wv[1];
      float wr[8] = {w01.x, w01.y, w01.z, w01.w, w23.x, w23.y, w23.z, w23.w};

      float part = 0.f;
#pragma unroll
      for (int r = 0; r < 8; ++r) part += fmaxf(d[r], 0.f) * wr[r];
      part += __shfl_xor(part, 16, 32);  // combine the two 8-feature halves

      if (lane < 16) out[p * N_PTS + n] = part + b2pk[p];
    }
  }
}

// ================= host glue =================
extern "C" void kernel_launch(void* const* d_in, const int* in_sizes, int n_in,
                              void* d_out, int out_size, void* d_ws, size_t ws_size,
                              hipStream_t stream) {
  const float* feats      = (const float*)d_in[0];
  const float* coords     = (const float*)d_in[1];
  const float* inst_feats = (const float*)d_in[2];
  const float* centers    = (const float*)d_in[3];
  const float* mask_w1    = (const float*)d_in[4];
  const float* mask_g1    = (const float*)d_in[5];
  const float* mask_b1    = (const float*)d_in[6];
  const float* mask_w2    = (const float*)d_in[7];
  const float* mask_g2    = (const float*)d_in[8];
  const float* mask_b2    = (const float*)d_in[9];
  const float* mask_w3    = (const float*)d_in[10];
  const float* mask_g3    = (const float*)d_in[11];
  const float* mask_b3    = (const float*)d_in[12];
  const float* mask_w_out = (const float*)d_in[13];
  const float* mask_b_out = (const float*)d_in[14];
  const float* emb_w      = (const float*)d_in[15];
  const float* emb_g      = (const float*)d_in[16];
  const float* emb_b      = (const float*)d_in[17];
  const float* ctrl_w     = (const float*)d_in[18];
  const float* ctrl_b     = (const float*)d_in[19];

  char* ws = (char*)d_ws;
  float*     stats = (float*)(ws + WS_STATS);
  _Float16*  w1A   = (_Float16*)(ws + WS_W1A);
  _Float16*  maskA = (_Float16*)(ws + WS_MASKA);
  _Float16*  woutA = (_Float16*)(ws + WS_WOUTA);
  float*     b1c   = (float*)(ws + WS_B1C);
  float*     w2pk  = (float*)(ws + WS_W2PK);
  float*     b2pk  = (float*)(ws + WS_B2PK);
  _Float16*  yA    = (_Float16*)(ws + WS_YA);
  _Float16*  yB    = (_Float16*)(ws + WS_YB);
  float*     out   = (float*)d_out;

  setup_kernel<<<1, 256, 0, stream>>>(inst_feats, centers, mask_w1, mask_w2, mask_w3,
                                      mask_w_out, emb_w, emb_g, emb_b, ctrl_w, ctrl_b,
                                      w1A, maskA, woutA, b1c, w2pk, b2pk, stats);

  const int LB = 360;  // layer-kernel blocks (8 waves each)
  mlp_layer_kernel<<<LB, 256, 0, stream>>>(feats, nullptr, maskA,
                                           nullptr, nullptr, nullptr,
                                           yA, stats + 0, 0);
  mlp_layer_kernel<<<LB, 256, 0, stream>>>(nullptr, yA, maskA + 1024,
                                           stats + 0, mask_g1, mask_b1,
                                           yB, stats + 64, 1);
  mlp_layer_kernel<<<LB, 256, 0, stream>>>(nullptr, yB, maskA + 2048,
                                           stats + 64, mask_g2, mask_b2,
                                           yA, stats + 128, 1);
  aug_kernel<<<LB, 256, 0, stream>>>(yA, woutA, stats + 128, mask_g3, mask_b3,
                                     mask_b_out, coords, yB);

  heavy_kernel<<<(TILES + 7) / 8, 256, 0, stream>>>(yB, w1A, b1c, w2pk, b2pk, out);
}
